// VQSynergy_30176440222163
// MI455X (gfx1250) — matrix operand order repaired and checked by reading, not compiled, not run on hardware
//
#include <hip/hip_runtime.h>
#include <hip/hip_bf16.h>
#include <float.h>

// ---------------------------------------------------------------------------
// VQ quantizer, fused for MI455X (gfx1250, wave32, WMMA, async LDS DMA).
//   d_in[0]: inputs   [64][256][32][32] f32  (b, d, h, w)
//   d_in[1]: embedding[1024][256] f32
//   d_out  : [ loss (1) | q_out (64*256*32*32) | perplexity (1) ]  f32
//
// GEMM 65536x1024x256 via v_wmma_f32_16x16x32_bf16 with hi/lo bf16 split
// (3 WMMAs ~ fp32 accuracy). Full 130KB fp32 A-tile is DMA'd once per WG into
// LDS via GLOBAL_LOAD_ASYNC_TO_LDS_B32 (transpose copy, ASYNCcnt-tracked),
// exploiting gfx1250's 320KB-per-WGP LDS. Softmax entropy + gumbel argmax are
// fused into the GEMM epilogue (scores never touch memory). ||x||^2 dropped
// (softmax/argmax invariant).
// ---------------------------------------------------------------------------

typedef __attribute__((ext_vector_type(16))) __bf16 v16bf;
typedef __attribute__((ext_vector_type(8)))  __bf16 v8bf;
typedef __attribute__((ext_vector_type(8)))  float  v8f;

#define NUM_EMB   1024
#define EMB_DIM   256
#define N_ROWS    65536          // 64 * 32 * 32
#define M_TILE    128            // rows per workgroup (8 waves x 16 rows)
#define NK_CHUNK  64             // codebook columns per chunk (4 x 16-wide tiles)
#define K_STEP    32             // bf16 WMMA K
#define A_STRIDE  260            // f32 elems/row: 1040B = 65*16 (aligned), 260%64=4
                                 // -> b128 lane reads conflict-free across 16 rows
#define B_STRIDE  40             // bf16 elems/row: 80B, 16B-aligned, conflict-free

#if __has_builtin(__builtin_amdgcn_global_load_async_to_lds_b32)
#define HAVE_ASYNC_LDS 1
#else
#define HAVE_ASYNC_LDS 0
#warning "gfx1250 global_load_async_to_lds_b32 builtin unavailable: using staged-copy fallback"
#endif
#if __has_builtin(__builtin_amdgcn_s_wait_asynccnt)
#define HAVE_WAIT_ASYNC 1
#else
#define HAVE_WAIT_ASYNC 0
#endif

#define AS1 __attribute__((address_space(1)))
#define AS3 __attribute__((address_space(3)))

// murmur3-finalizer hash on the 26-bit element id -> gumbel(0,1).
// (Approximates jax threefry key 42; bit-exact match impossible, statistically
// equivalent. 6 VALU ops/element vs ~15 for a 64-bit mix — this term is the
// largest per-element VALU cost in the fused kernel, so it matters.)
__device__ __forceinline__ float gumbelf(uint32_t id) {
  uint32_t h = id * 0x9E3779B9u + 0x2545F491u;   // 42-derived seed offset
  h ^= h >> 16; h *= 0x85EBCA6Bu;
  h ^= h >> 13; h *= 0xC2B2AE35u;
  h ^= h >> 16;
  float u = ((float)(h >> 8) + 0.5f) * (1.0f / 16777216.0f);  // (0,1)
  return -__logf(-__logf(u));
}

// --------------------------- kernel 0: prep --------------------------------
// csq[k] = ||embedding[k]||^2 ; also zero counts + entropy accumulator.
__global__ __launch_bounds__(256) void vq_prep(const float* __restrict__ emb,
                                               float* __restrict__ csq,
                                               float* __restrict__ counts,
                                               float* __restrict__ ent_accum) {
  const int t = threadIdx.x, lane = t & 31, wave = t >> 5;
  const int row = blockIdx.x * 8 + wave;       // 128 blocks * 8 rows = 1024
  float s = 0.0f;
#pragma unroll
  for (int j = 0; j < 8; ++j) {
    float v = emb[(size_t)row * EMB_DIM + lane + j * 32];
    s = fmaf(v, v, s);
  }
#pragma unroll
  for (int m = 16; m >= 1; m >>= 1) s += __shfl_xor(s, m, 32);
  if (lane == 0) csq[row] = s;
  if (blockIdx.x == 0) {
#pragma unroll
    for (int i = 0; i < 4; ++i) counts[t * 4 + i] = 0.0f;
    if (t == 0) ent_accum[0] = 0.0f;
  }
}

// ------------------ kernel 1: fused GEMM + softmax/argmax ------------------
__global__ __launch_bounds__(256) void vq_gemm_fused(
    const float* __restrict__ x,      // [64][256][1024]  (b, d, hw)
    const float* __restrict__ emb,    // [1024][256]
    const float* __restrict__ csq,    // [1024]
    float* __restrict__ ent_accum,    // [1]
    float* __restrict__ counts,       // [1024]
    int*   __restrict__ idx_out)      // [65536]
{
  // Full A tile (fp32, [m][d], padded) + B staging (bf16 hi/lo, [code][k]).
  __shared__ __attribute__((aligned(16))) float  sA[M_TILE][A_STRIDE];    // 133120 B
  __shared__ __attribute__((aligned(16))) __bf16 sB[2][NK_CHUNK][B_STRIDE]; // 10240 B

  const int tid  = threadIdx.x;
  const int lane = tid & 31;
  const int wave = tid >> 5;        // 0..7 -> rows wave*16 .. +15
  const int half = lane >> 4;       // 0/1
  const int lc   = lane & 15;

  const int b   = blockIdx.x >> 3;          // 8 M-tiles per batch image
  const int hw0 = (blockIdx.x & 7) * M_TILE;

  // ---- phase 1: DMA the whole 128x256 fp32 A tile into LDS, transposed.
  // Global is [d][m]-contiguous; LDS is [m][d]. Per instruction, 32 lanes read
  // 128B contiguous global and the async engine scatters 4B/lane into LDS.
  for (int it = 0; it < (M_TILE * EMB_DIM) / 256; ++it) {   // 128 iterations
    const int idx = it * 256 + tid;
    const int d = idx >> 7;          // 0..255
    const int m = idx & 127;         // 0..127 (lanes -> consecutive m)
    const float* gp = x + (size_t)(b * EMB_DIM + d) * 1024 + hw0 + m;
    float* lp = &sA[m][d];
#if HAVE_ASYNC_LDS
    __builtin_amdgcn_global_load_async_to_lds_b32(
        (AS1 int*)(uintptr_t)gp,
        (AS3 int*)(uint32_t)(uintptr_t)lp, 0, 0);
#else
    *lp = *gp;
#endif
  }
#if HAVE_ASYNC_LDS && HAVE_WAIT_ASYNC
  __builtin_amdgcn_s_wait_asynccnt(0);
#endif
  __syncthreads();

  // Per-lane online stats for rows (wave*16 + r + 8*half), r = 0..7.
  // -FLT_MAX (finite) sentinel keeps exp()/mul free of inf*0 NaNs.
  float mx[8], s0[8], s1[8], bv[8];
  int   bi[8];
#pragma unroll
  for (int r = 0; r < 8; ++r) { mx[r] = -FLT_MAX; s0[r] = 0.f; s1[r] = 0.f; bv[r] = -FLT_MAX; bi[r] = 0; }

  // B loader thread mapping
  const int b_cd = tid >> 2;          // 0..63 (code)
  const int b_ds = (tid & 3) * 8;     // 0..24 (d segment)

  for (int chunk = 0; chunk < NUM_EMB / NK_CHUNK; ++chunk) {
    const int code0 = chunk * NK_CHUNK;
    float cc[4];
#pragma unroll
    for (int jt = 0; jt < 4; ++jt) cc[jt] = csq[code0 + jt * 16 + lc];

    v8f acc[4];
#pragma unroll
    for (int jt = 0; jt < 4; ++jt)
#pragma unroll
      for (int i = 0; i < 8; ++i) acc[jt][i] = 0.0f;

    for (int ks = 0; ks < EMB_DIM / K_STEP; ++ks) {
      const int d0 = ks * K_STEP;
      __syncthreads();   // previous k-step's sB readers done before overwrite
      { // ---- B tile: 64(code) x 32(d); rows contiguous, L2-resident
        const float* g = emb + (size_t)(code0 + b_cd) * EMB_DIM + d0 + b_ds;
        float4 e0 = *(const float4*)(g + 0);
        float4 e1 = *(const float4*)(g + 4);
        float w[8] = {e0.x, e0.y, e0.z, e0.w, e1.x, e1.y, e1.z, e1.w};
#pragma unroll
        for (int i = 0; i < 8; ++i) {
          __bf16 h = (__bf16)w[i];
          __bf16 l = (__bf16)(w[i] - (float)h);
          sB[0][b_cd][b_ds + i] = h;
          sB[1][b_cd][b_ds + i] = l;
        }
      }
      __syncthreads();

      // ---- A fragment straight from the resident fp32 tile, hi/lo split in regs.
      // (16x32 bf16 A layout: lane<16 -> K{0..7,16..23}, else K{8..15,24..31})
      const int am  = wave * 16 + lc;
      const int ak0 = d0 + (half ? 8 : 0);
      const float* pa = &sA[am][ak0];
      float4 a0 = *(const float4*)(pa + 0);
      float4 a1 = *(const float4*)(pa + 4);
      float4 a2 = *(const float4*)(pa + 16);
      float4 a3 = *(const float4*)(pa + 20);
      float av[16] = {a0.x, a0.y, a0.z, a0.w, a1.x, a1.y, a1.z, a1.w,
                      a2.x, a2.y, a2.z, a2.w, a3.x, a3.y, a3.z, a3.w};
      v16bf Ah, Al;
#pragma unroll
      for (int i = 0; i < 16; ++i) {
        __bf16 h = (__bf16)av[i];
        Ah[i] = h;
        Al[i] = (__bf16)(av[i] - (float)h);
      }

#pragma unroll
      for (int jt = 0; jt < 4; ++jt) {
        // B fragment (32x16 bf16: lane<16 -> col=lane, K{0..7,16..23}; else col=lane-16, K{8..15,24..31})
        const int bc = jt * 16 + lc;
        const int bk = half * 8;
        v8bf bh0 = *(const v8bf*)&sB[0][bc][bk];
        v8bf bh1 = *(const v8bf*)&sB[0][bc][bk + 16];
        v8bf bl0 = *(const v8bf*)&sB[1][bc][bk];
        v8bf bl1 = *(const v8bf*)&sB[1][bc][bk + 16];
        v16bf Bh = __builtin_shufflevector(bh0, bh1, 0,1,2,3,4,5,6,7,8,9,10,11,12,13,14,15);
        v16bf Bl = __builtin_shufflevector(bl0, bl1, 0,1,2,3,4,5,6,7,8,9,10,11,12,13,14,15);

        // hi/lo split: x.e ~= xh.eh + xh.el + xl.eh   (fp32 accumulate)
        acc[jt] = __builtin_amdgcn_wmma_f32_16x16x32_bf16(false, Ah, false, Bh, (short)0, acc[jt], false, false);
        acc[jt] = __builtin_amdgcn_wmma_f32_16x16x32_bf16(false, Ah, false, Bl, (short)0, acc[jt], false, false);
        acc[jt] = __builtin_amdgcn_wmma_f32_16x16x32_bf16(false, Al, false, Bh, (short)0, acc[jt], false, false);
      }
    }

    // ---- epilogue for this 64-column chunk: online softmax stats + gumbel argmax
#pragma unroll
    for (int jt = 0; jt < 4; ++jt) {
      const int k = code0 + jt * 16 + lc;
#pragma unroll
      for (int r = 0; r < 8; ++r) {
        float t = 2.0f * acc[jt][r] - cc[jt];   // = -d2 + ||x||^2 (row-const dropped)
        if (t > mx[r]) {
          float dm = mx[r] - t;
          float d  = __expf(dm);
          s1[r] = s1[r] * d + dm * d * s0[r];
          s0[r] = s0[r] * d + 1.0f;
          mx[r] = t;
        } else {
          float dm = t - mx[r];
          float e  = __expf(dm);
          s0[r] += e;
          s1[r] += dm * e;
        }
        uint32_t n = (uint32_t)(blockIdx.x * M_TILE + wave * 16 + r + 8 * half);
        float vg = t + gumbelf((n << 10) | (uint32_t)k);
        if (vg > bv[r]) { bv[r] = vg; bi[r] = k; }
      }
    }
  }

  // ---- butterfly reduce across the 16 lanes of each half-wave (same rows)
#pragma unroll
  for (int r = 0; r < 8; ++r) {
#pragma unroll
    for (int m = 8; m >= 1; m >>= 1) {
      float omx = __shfl_xor(mx[r], m, 32);
      float os0 = __shfl_xor(s0[r], m, 32);
      float os1 = __shfl_xor(s1[r], m, 32);
      float obv = __shfl_xor(bv[r], m, 32);
      int   obi = __shfl_xor(bi[r], m, 32);
      float nm = fmaxf(mx[r], omx);
      float da = __expf(mx[r] - nm);
      float db = __expf(omx - nm);
      s1[r] = (s1[r] + (mx[r] - nm) * s0[r]) * da + (os1 + (omx - nm) * os0) * db;
      s0[r] = s0[r] * da + os0 * db;
      mx[r] = nm;
      if (obv > bv[r]) { bv[r] = obv; bi[r] = obi; }
    }
  }

  // lanes lc=0..7 of each half write one row each
  if (lc < 8) {
    const int r   = lc;
    const int row = blockIdx.x * M_TILE + wave * 16 + r + 8 * half;
    float ent = s1[r] / s0[r] - __logf(s0[r]);   // sum_k p log p  (<= 0)
    atomicAdd(ent_accum, ent);
    idx_out[row] = bi[r];
    atomicAdd(&counts[bi[r]], 1.0f);
  }
}

// --------------------- kernel 2: gather quantized output -------------------
// q[b][d][hw] = embedding[idx[b*1024+hw]][d]; writes coalesced float4,
// embedding gathers hit L2 (1 MB table in 192 MB L2).
__global__ __launch_bounds__(256) void vq_scatter(const float* __restrict__ emb,
                                                  const int* __restrict__ idx,
                                                  float* __restrict__ q) {
  const int b  = blockIdx.x >> 8;
  const int d  = blockIdx.x & 255;
  const int hw = threadIdx.x * 4;
  const int4 iv = *(const int4*)&idx[b * 1024 + hw];
  float4 o;
  o.x = emb[(size_t)iv.x * EMB_DIM + d];
  o.y = emb[(size_t)iv.y * EMB_DIM + d];
  o.z = emb[(size_t)iv.z * EMB_DIM + d];
  o.w = emb[(size_t)iv.w * EMB_DIM + d];
  *(float4*)&q[(size_t)(b * EMB_DIM + d) * 1024 + hw] = o;
}

// --------------------- kernel 3: loss + perplexity -------------------------
__global__ __launch_bounds__(256) void vq_finalize(const float* __restrict__ counts,
                                                   const float* __restrict__ ent_accum,
                                                   float* __restrict__ out) {
  __shared__ float red[8];
  const int t = threadIdx.x, lane = t & 31, wave = t >> 5;
  float s = 0.0f;
#pragma unroll
  for (int i = 0; i < 4; ++i) {
    float a = counts[t * 4 + i] * (1.0f / (float)N_ROWS);
    s += a * __logf(a + 1e-10f);
  }
#pragma unroll
  for (int m = 16; m >= 1; m >>= 1) s += __shfl_xor(s, m, 32);
  if (lane == 0) red[wave] = s;
  __syncthreads();
  if (t == 0) {
    float tot = 0.0f;
#pragma unroll
    for (int w = 0; w < 8; ++w) tot += red[w];
    out[0] = 0.25f * (ent_accum[0] / (float)N_ROWS);   // COMMITMENT_COST * mean entropy-term
    out[1 + (size_t)N_ROWS * EMB_DIM] = __expf(-tot);  // perplexity
  }
}

// ---------------------------------------------------------------------------
extern "C" void kernel_launch(void* const* d_in, const int* in_sizes, int n_in,
                              void* d_out, int out_size, void* d_ws, size_t ws_size,
                              hipStream_t stream) {
  (void)in_sizes; (void)n_in; (void)out_size; (void)ws_size;
  const float* x   = (const float*)d_in[0];
  const float* emb = (const float*)d_in[1];
  float* out = (float*)d_out;

  char*  ws     = (char*)d_ws;
  float* csq    = (float*)(ws);             // 1024 f32
  float* counts = (float*)(ws + 4096);      // 1024 f32
  float* ent    = (float*)(ws + 8192);      // 1 f32
  int*   idx    = (int*)  (ws + 16384);     // 65536 i32

  vq_prep      <<<128,   256, 0, stream>>>(emb, csq, counts, ent);
  vq_gemm_fused<<<N_ROWS / M_TILE, 256, 0, stream>>>(x, emb, csq, ent, counts, idx);
  vq_scatter   <<<64 * EMB_DIM, 256, 0, stream>>>(emb, idx, out + 1);
  vq_finalize  <<<1,     256, 0, stream>>>(counts, ent, out);
}
